// ExpertRouter_11330123727025
// MI455X (gfx1250) — matrix operand order repaired and checked by reading, not compiled, not run on hardware
//
#include <hip/hip_runtime.h>

#define D_MODEL 4096
#define NUM_EXPERTS 64
#define TOKENS_PER_BLOCK 128
#define KC 32
#define KS 40  // LDS row stride in bf16 elems (80B = 20 dwords; gcd(20,64)=4 -> conflict-free A reads)

typedef __attribute__((ext_vector_type(16))) __bf16 v16bf;
typedef __attribute__((ext_vector_type(8)))  float  v8f;

struct alignas(16) US8 { unsigned short s[8]; };

// Pack two f32 -> two bf16 (RNE). The half-merge is a single v_perm_b32:
// result = { lo16 = bytes[3:2] of rounded a, hi16 = bytes[3:2] of rounded b }.
static __device__ inline unsigned pk2(float a, float b) {
  unsigned ua = __builtin_bit_cast(unsigned, a);
  unsigned ub = __builtin_bit_cast(unsigned, b);
  ua += 0x7FFFu + ((ua >> 16) & 1u);
  ub += 0x7FFFu + ((ub >> 16) & 1u);
  return __builtin_amdgcn_perm(ub, ua, 0x07060302u);
}

static __device__ inline v16bf lds_frag(const unsigned short* p0,
                                        const unsigned short* p1) {
  US8 lo = *(const US8*)p0;  // ds_load_b128
  US8 hi = *(const US8*)p1;  // ds_load_b128
  v16bf r;
#pragma unroll
  for (int i = 0; i < 8; ++i) {
    r[i]     = __builtin_bit_cast(__bf16, lo.s[i]);
    r[i + 8] = __builtin_bit_cast(__bf16, hi.s[i]);
  }
  return r;
}

__global__ __launch_bounds__(256) void router_gemm_kernel(
    const float* __restrict__ xg, const float* __restrict__ wg,
    const float* __restrict__ bg, float* __restrict__ logits, int tokens) {
  __shared__ unsigned short XS[TOKENS_PER_BLOCK * KS];  // 10 KB
  __shared__ unsigned short WS[NUM_EXPERTS   * KS];     // 5 KB

  const int tid     = threadIdx.x;
  const int lane    = tid & 31;        // wave32
  const int wid     = tid >> 5;        // 8 waves
  const int tokBase = blockIdx.x * TOKENS_PER_BLOCK;

  v8f acc0 = {}, acc1 = {}, acc2 = {}, acc3 = {};

  for (int k0 = 0; k0 < D_MODEL; k0 += KC) {
    // ---- stage x tile (128 tokens x 32 k) as bf16, coalesced float4 loads ----
#pragma unroll
    for (int i = 0; i < 4; ++i) {
      int s  = tid + i * 256;          // float4 slot 0..1023
      int r  = s >> 3;                 // token row 0..127
      int kk = (s & 7) << 2;           // 0..28
      float4 v = *(const float4*)(xg + (size_t)(tokBase + r) * D_MODEL + k0 + kk);
      uint2 pk = make_uint2(pk2(v.x, v.y), pk2(v.z, v.w));
      *(uint2*)(&XS[r * KS + kk]) = pk;  // ds_store_b64, 8B-aligned
    }
    // ---- stage w chunk transposed: each thread packs a 2(k) x 4(e) sub-block ----
    {
      int kk = (tid >> 4) << 1;        // 0,2,...,30
      int e4 = (tid & 15) << 2;        // 0,4,...,60
      const float* wp = wg + (size_t)(k0 + kk) * NUM_EXPERTS + e4;
      float4 w0 = *(const float4*)(wp);                 // row k0+kk
      float4 w1 = *(const float4*)(wp + NUM_EXPERTS);   // row k0+kk+1
      *(unsigned*)(&WS[(e4 + 0) * KS + kk]) = pk2(w0.x, w1.x);  // ds_store_b32
      *(unsigned*)(&WS[(e4 + 1) * KS + kk]) = pk2(w0.y, w1.y);
      *(unsigned*)(&WS[(e4 + 2) * KS + kk]) = pk2(w0.z, w1.z);
      *(unsigned*)(&WS[(e4 + 3) * KS + kk]) = pk2(w0.w, w1.w);
    }
    // prefetch next x chunk (one 128B line per token row)
    if (k0 + KC < D_MODEL && tid < TOKENS_PER_BLOCK) {
      __builtin_prefetch(xg + (size_t)(tokBase + tid) * D_MODEL + k0 + KC, 0, 3);
    }
    __syncthreads();

    // ---- A fragment: 16 tokens x 32 k (contiguous k = 16*(lane/16) + s, matches B) ----
    const unsigned short* xrow =
        &XS[(wid * 16 + (lane & 15)) * KS + ((lane < 16) ? 0 : 16)];
    v16bf a = lds_frag(xrow, xrow + 8);

    // ---- 4 B fragments (32 k x 16 experts each) + WMMA ----
    {
      const unsigned short* wrow = &WS[(0 * 16 + (lane & 15)) * KS + ((lane < 16) ? 0 : 16)];
      acc0 = __builtin_amdgcn_wmma_f32_16x16x32_bf16(false, a, false,
              lds_frag(wrow, wrow + 8), (short)0, acc0, false, false);
    }
    {
      const unsigned short* wrow = &WS[(1 * 16 + (lane & 15)) * KS + ((lane < 16) ? 0 : 16)];
      acc1 = __builtin_amdgcn_wmma_f32_16x16x32_bf16(false, a, false,
              lds_frag(wrow, wrow + 8), (short)0, acc1, false, false);
    }
    {
      const unsigned short* wrow = &WS[(2 * 16 + (lane & 15)) * KS + ((lane < 16) ? 0 : 16)];
      acc2 = __builtin_amdgcn_wmma_f32_16x16x32_bf16(false, a, false,
              lds_frag(wrow, wrow + 8), (short)0, acc2, false, false);
    }
    {
      const unsigned short* wrow = &WS[(3 * 16 + (lane & 15)) * KS + ((lane < 16) ? 0 : 16)];
      acc3 = __builtin_amdgcn_wmma_f32_16x16x32_bf16(false, a, false,
              lds_frag(wrow, wrow + 8), (short)0, acc3, false, false);
    }
    __syncthreads();
  }

  // ---- epilogue: bias add + store (C layout: N=lane%16, M=vgpr + 8*(lane>=16)) ----
  const int eb    = lane & 15;
  const int mBase = tokBase + wid * 16 + ((lane < 16) ? 0 : 8);
  v8f accs[4] = {acc0, acc1, acc2, acc3};
#pragma unroll
  for (int nb = 0; nb < 4; ++nb) {
    const int e = nb * 16 + eb;
    const float bias = bg[e];
#pragma unroll
    for (int v = 0; v < 8; ++v) {
      logits[(size_t)(mBase + v) * NUM_EXPERTS + e] = accs[nb][v] + bias;
    }
  }
}

__global__ __launch_bounds__(256) void router_topk_kernel(
    const float* __restrict__ logits, int* __restrict__ idx,
    float* __restrict__ gates, int tokens) {
  int t = blockIdx.x * blockDim.x + threadIdx.x;
  if (t >= tokens) return;
  const float* row = logits + (size_t)t * NUM_EXPERTS;
  float b0 = -3.402823e38f; int i0 = 0;
  float b1 = -3.402823e38f; int i1 = 0;
#pragma unroll
  for (int e = 0; e < NUM_EXPERTS; e += 4) {
    float4 v4 = *(const float4*)(row + e);
    float vv[4] = {v4.x, v4.y, v4.z, v4.w};
#pragma unroll
    for (int j = 0; j < 4; ++j) {
      float val = vv[j];
      if (val > b0)      { b1 = b0; i1 = i0; b0 = val; i0 = e + j; }
      else if (val > b1) { b1 = val; i1 = e + j; }
    }
  }
  // softmax over the two selected logits (descending order)
  float e1  = expf(b1 - b0);
  float inv = 1.0f / (1.0f + e1);
  idx[t * 2]       = i0;
  idx[t * 2 + 1]   = i1;
  gates[t * 2]     = inv;
  gates[t * 2 + 1] = e1 * inv;
}

extern "C" void kernel_launch(void* const* d_in, const int* in_sizes, int n_in,
                              void* d_out, int out_size, void* d_ws, size_t ws_size,
                              hipStream_t stream) {
  (void)n_in; (void)out_size; (void)d_ws; (void)ws_size;
  const float* x = (const float*)d_in[0];
  const float* w = (const float*)d_in[1];
  const float* b = (const float*)d_in[2];
  const int tokens = in_sizes[0] / D_MODEL;  // 16384

  float* logits = (float*)d_out;
  int*   idx    = (int*)(logits + (size_t)tokens * NUM_EXPERTS);
  float* gates  = (float*)(idx + (size_t)tokens * 2);

  router_gemm_kernel<<<dim3(tokens / TOKENS_PER_BLOCK), 256, 0, stream>>>(
      x, w, b, logits, tokens);
  router_topk_kernel<<<dim3((tokens + 255) / 256), 256, 0, stream>>>(
      logits, idx, gates, tokens);
}
